// Int4PackedLinear_73521250173343
// MI455X (gfx1250) — 8-bit matrix operand order repaired and checked by reading, compiled, not run on hardware
//
#include <hip/hip_runtime.h>
#include <hip/hip_bf16.h>
#include <cstdint>

typedef __attribute__((ext_vector_type(8))) int v8i;

namespace {
constexpr int IN_F  = 4096;
constexpr int OUT_F = 4096;
constexpr int BATCH = 4;
constexpr int SEQ   = 2048;
constexpr int M = BATCH * SEQ;
constexpr int N = OUT_F;
constexpr int K = IN_F;

constexpr int WAVE_M = 32;
constexpr int WAVE_N = 64;
constexpr int BLK_M  = 128;
constexpr int BLK_N  = 128;
}

__device__ __forceinline__ float bf16_rne(float f) {
    unsigned int u  = __float_as_uint(f);
    unsigned int lsb = (u >> 16) & 1u;
    u += 0x7FFFu + lsb;
    u &= 0xFFFF0000u;
    return __uint_as_float(u);
}

__device__ __forceinline__ int clamp127(int v) {
    return v < -127 ? -127 : (v > 127 ? 127 : v);
}

__global__ void zero_amax_kernel(unsigned int* __restrict__ amax_bits) {
    if (threadIdx.x == 0 && blockIdx.x == 0) *amax_bits = 0u;
}

__global__ void amax_reduce_kernel(const float4* __restrict__ x4, int n4,
                                   unsigned int* __restrict__ amax_bits) {
    __shared__ float smax[256];
    float m = 0.0f;
    for (int i = blockIdx.x * blockDim.x + threadIdx.x; i < n4;
         i += gridDim.x * blockDim.x) {
        float4 v = x4[i];
        m = fmaxf(m, fmaxf(fmaxf(fabsf(v.x), fabsf(v.y)),
                           fmaxf(fabsf(v.z), fabsf(v.w))));
    }
    smax[threadIdx.x] = m;
    __syncthreads();
    for (int s = 128; s > 0; s >>= 1) {
        if ((int)threadIdx.x < s)
            smax[threadIdx.x] = fmaxf(smax[threadIdx.x], smax[threadIdx.x + s]);
        __syncthreads();
    }
    if (threadIdx.x == 0)
        atomicMax(amax_bits, __float_as_uint(smax[0]));
}

__global__ void quantize_x_kernel(const float4* __restrict__ x4,
                                  const unsigned int* __restrict__ amax_bits,
                                  int* __restrict__ x8, int n4) {
    int i = blockIdx.x * blockDim.x + threadIdx.x;
    if (i >= n4) return;
    float amax = __uint_as_float(*amax_bits);
    float xs   = (amax == 0.0f) ? 1.0f : amax / 127.0f;
    float4 v = x4[i];
    int q0 = clamp127(__float2int_rn(v.x / xs));
    int q1 = clamp127(__float2int_rn(v.y / xs));
    int q2 = clamp127(__float2int_rn(v.z / xs));
    int q3 = clamp127(__float2int_rn(v.w / xs));
    x8[i] = (q0 & 255) | ((q1 & 255) << 8) | ((q2 & 255) << 16) | ((q3 & 255) << 24);
}

__global__ void pack_w_kernel(const int4* __restrict__ w4,
                              int* __restrict__ w8, int n4) {
    int i = blockIdx.x * blockDim.x + threadIdx.x;
    if (i >= n4) return;
    int4 v = w4[i];
    w8[i] = (v.x & 255) | ((v.y & 255) << 8) | ((v.z & 255) << 16) | ((v.w & 255) << 24);
}

__device__ __forceinline__ v8i load_a_frag(const int8_t* p) {
    int2 d0 = *(const int2*)(p);
    int2 d1 = *(const int2*)(p + 8);
    int2 d2 = *(const int2*)(p + 32);
    int2 d3 = *(const int2*)(p + 40);
    v8i r;
    r[0] = d0.x; r[1] = d0.y; r[2] = d1.x; r[3] = d1.y;
    r[4] = d2.x; r[5] = d2.y; r[6] = d3.x; r[7] = d3.y;
    return r;
}

__device__ __forceinline__ v8i load_b_frag(const int8_t* p) {
    int4 e0 = *(const int4*)(p);
    int4 e1 = *(const int4*)(p + 32);
    v8i r;
    r[0] = e0.x; r[1] = e0.y; r[2] = e0.z; r[3] = e0.w;
    r[4] = e1.x; r[5] = e1.y; r[6] = e1.z; r[7] = e1.w;
    return r;
}

__device__ __forceinline__ void wmma_group(const v8i a[2], const v8i b[4],
                                           v8i acc[2][4]) {
#pragma unroll
    for (int mt = 0; mt < 2; ++mt)
#pragma unroll
        for (int nt = 0; nt < 4; ++nt)
            acc[mt][nt] = __builtin_amdgcn_wmma_i32_16x16x64_iu8(
                 true, a[mt],  true, b[nt],
                acc[mt][nt],  false,  false);
}

__global__ __launch_bounds__(256)
void int8_gemm_wmma_kernel(const int8_t* __restrict__ A8,
                           const int8_t* __restrict__ W8,
                           const unsigned int* __restrict__ amax_bits,
                           const float* __restrict__ w_scale_p,
                           const float* __restrict__ bias,
                           float* __restrict__ out) {
    const int lane = threadIdx.x & 31;
    const int wid  = threadIdx.x >> 5;
    const int h    = lane >> 4;
    const int ln   = lane & 15;

    const int wm = wid & 3;
    const int wn = wid >> 2;
    const int mbase = blockIdx.y * BLK_M + wm * WAVE_M;
    const int nbase = blockIdx.x * BLK_N + wn * WAVE_N;

    const int8_t* pa0 = A8 + (size_t)(mbase + ln) * K + 16 * h;
    const int8_t* pa1 = pa0 + (size_t)16 * K;
    const int8_t* pb0 = W8 + (size_t)(nbase + 0  + ln) * K + 16 * h;
    const int8_t* pb1 = W8 + (size_t)(nbase + 16 + ln) * K + 16 * h;
    const int8_t* pb2 = W8 + (size_t)(nbase + 32 + ln) * K + 16 * h;
    const int8_t* pb3 = W8 + (size_t)(nbase + 48 + ln) * K + 16 * h;

    v8i acc[2][4] = {};
    v8i a0[2], b0[4], a1[2], b1[4];

    a0[0] = load_a_frag(pa0);
    a0[1] = load_a_frag(pa1);
    b0[0] = load_b_frag(pb0);
    b0[1] = load_b_frag(pb1);
    b0[2] = load_b_frag(pb2);
    b0[3] = load_b_frag(pb3);
    pa0 += 64; pa1 += 64; pb0 += 64; pb1 += 64; pb2 += 64; pb3 += 64;

#pragma unroll 1
    for (int it = 0; it < (K / 64 - 2) / 2; ++it) {
        a1[0] = load_a_frag(pa0);
        a1[1] = load_a_frag(pa1);
        b1[0] = load_b_frag(pb0);
        b1[1] = load_b_frag(pb1);
        b1[2] = load_b_frag(pb2);
        b1[3] = load_b_frag(pb3);
        wmma_group(a0, b0, acc);

        a0[0] = load_a_frag(pa0 + 64);
        a0[1] = load_a_frag(pa1 + 64);
        b0[0] = load_b_frag(pb0 + 64);
        b0[1] = load_b_frag(pb1 + 64);
        b0[2] = load_b_frag(pb2 + 64);
        b0[3] = load_b_frag(pb3 + 64);
        wmma_group(a1, b1, acc);

        pa0 += 128; pa1 += 128; pb0 += 128; pb1 += 128; pb2 += 128; pb3 += 128;
    }

    a1[0] = load_a_frag(pa0);
    a1[1] = load_a_frag(pa1);
    b1[0] = load_b_frag(pb0);
    b1[1] = load_b_frag(pb1);
    b1[2] = load_b_frag(pb2);
    b1[3] = load_b_frag(pb3);
    wmma_group(a0, b0, acc);
    wmma_group(a1, b1, acc);

    float amax = __uint_as_float(*amax_bits);
    float xs   = (amax == 0.0f) ? 1.0f : amax / 127.0f;
    float cs_b = bf16_rne(xs * w_scale_p[0]);

#pragma unroll
    for (int nt = 0; nt < 4; ++nt) {
        const int n = nbase + nt * 16 + ln;
        const float bn = bias[n];
#pragma unroll
        for (int mt = 0; mt < 2; ++mt) {
#pragma unroll
            for (int r = 0; r < 8; ++r) {
                const int m = mbase + mt * 16 + 8 * h + r;
                float prod = bf16_rne(bf16_rne((float)acc[mt][nt][r]) * cs_b);
                out[(size_t)m * N + n] = prod + bn;
            }
        }
    }
}

extern "C" void kernel_launch(void* const* d_in, const int* in_sizes, int n_in,
                              void* d_out, int out_size, void* d_ws, size_t ws_size,
                              hipStream_t stream) {
    const float* x       = (const float*)d_in[0];
    const int*   w_int   = (const int*)d_in[1];
    const float* w_scale = (const float*)d_in[2];
    const float* bias    = (const float*)d_in[3];
    float*       out     = (float*)d_out;

    unsigned int* amax_bits = (unsigned int*)d_ws;
    int8_t* x8 = (int8_t*)d_ws + 256;
    int8_t* w8 = x8 + (size_t)M * K;

    zero_amax_kernel<<<1, 64, 0, stream>>>(amax_bits);

    const int n4x = M * K / 4;
    amax_reduce_kernel<<<2048, 256, 0, stream>>>((const float4*)x, n4x, amax_bits);
    quantize_x_kernel<<<(n4x + 255) / 256, 256, 0, stream>>>(
        (const float4*)x, amax_bits, (int*)x8, n4x);

    const int n4w = N * K / 4;
    pack_w_kernel<<<(n4w + 255) / 256, 256, 0, stream>>>(
        (const int4*)w_int, (int*)w8, n4w);

    dim3 grid(N / BLK_N, M / BLK_M);
    int8_gemm_wmma_kernel<<<grid, 256, 0, stream>>>(x8, w8, amax_bits,
                                                    w_scale, bias, out);
}
